// GQAHead_18459769438651
// MI455X (gfx1250) — matrix operand order repaired and checked by reading, not compiled, not run on hardware
//
#include <hip/hip_runtime.h>

// ---------------------------------------------------------------------------
// GQA head: out = softmax(causal(RoPE(x@Wq^T) @ RoPE(key)^T / sqrt(128))) @ value
// B=4, T=4096, C=2048, HEAD=128.  bf16 WMMA matmuls, fp32 accumulation,
// flash-attention, TDM (tensor_load_to_lds) for Q/K tile staging.
// ---------------------------------------------------------------------------

#define T_SEQ 4096
#define C_DIM 2048
#define HEAD  128
#define BATCH 4

typedef __bf16 bf16_t;
typedef __attribute__((ext_vector_type(16))) __bf16 v16bf;
typedef __attribute__((ext_vector_type(8)))  __bf16 v8bf;
typedef __attribute__((ext_vector_type(8)))  float  v8f;
typedef __attribute__((ext_vector_type(4)))  unsigned uint32x4;
typedef __attribute__((ext_vector_type(8)))  int      int32x8;
typedef __attribute__((ext_vector_type(4)))  int      int32x4;

__device__ __forceinline__ bf16_t f2bf(float f) {
  unsigned u = __builtin_bit_cast(unsigned, f);
  unsigned r = (u + 0x7FFFu + ((u >> 16) & 1u)) >> 16;  // round-to-nearest-even
  return __builtin_bit_cast(bf16_t, (unsigned short)r);
}

// A-matrix fragment (16x32 bf16, M x K): lane holds row M=lane%16.
// elements 0..7 -> K = khalf..khalf+7, elements 8..15 -> K = khalf+16..+23,
// khalf = 0 (lanes 0-15) or 8 (lanes 16-31).   [ISA 7.12.2]
__device__ __forceinline__ v16bf load_fragA(const bf16_t* tile, int row0,
                                            int col0, int stride, int lane) {
  int r = row0 + (lane & 15);
  int khalf = (lane < 16) ? 0 : 8;
  const bf16_t* p = tile + r * stride + col0 + khalf;
  v8bf lo = *(const v8bf*)p;
  v8bf hi = *(const v8bf*)(p + 16);
  v16bf o;
#pragma unroll
  for (int i = 0; i < 8; ++i) { o[i] = lo[i]; o[i + 8] = hi[i]; }
  return o;
}

// B-matrix fragment (32x16 bf16, K x N), stored transposed in LDS as tile[n][k]:
// lane holds column N=lane%16; elements 0..7 -> K = col0+khalf..+7,
// elements 8..15 -> K = col0+khalf+16..+23, khalf = 0 / 8 (same as A).  [ISA 7.12.4 B layout]
__device__ __forceinline__ v16bf load_fragB(const bf16_t* tile, int row0,
                                            int col0, int stride, int lane) {
  int n = row0 + (lane & 15);
  int kb = col0 + ((lane < 16) ? 0 : 8);
  const bf16_t* p = tile + n * stride + kb;
  v8bf lo = *(const v8bf*)p;
  v8bf hi = *(const v8bf*)(p + 16);
  v16bf o;
#pragma unroll
  for (int i = 0; i < 8; ++i) { o[i] = lo[i]; o[i + 8] = hi[i]; }
  return o;
}

__device__ __forceinline__ v8f wmma_bf16(v16bf a, v16bf b, v8f c) {
  return __builtin_amdgcn_wmma_f32_16x16x32_bf16(false, a, false, b,
                                                 (short)0, c, false, false);
}

// LDS byte offset of a shared-memory pointer (flat LDS addr low 32 bits).
__device__ __forceinline__ unsigned lds_off(const void* p) {
  return (unsigned)(size_t)p;
}

// TDM: DMA a 2D bf16 tile (tile_rows x 128 elems, row-major, dim0 stride =
// row_stride elems) from global memory into LDS at lds_addr, inserting 4
// DWORDs (8 bf16) of padding after every 64 DWORDs (256B = one 128-elem row)
// -> effective LDS row stride = 136 bf16.   [ISA 8.3/8.4 D# descriptor]
__device__ __forceinline__ void tdm_load_tile_bf16(
    unsigned lds_addr, const bf16_t* gptr, unsigned tile_rows,
    unsigned tensor_rows, unsigned row_stride) {
  unsigned long long ga = (unsigned long long)(size_t)gptr;
  uint32x4 g0;
  g0[0] = 1u;                                           // count=1, user mode
  g0[1] = lds_addr;                                     // lds_addr
  g0[2] = (unsigned)ga;                                 // global_addr[31:0]
  g0[3] = (unsigned)((ga >> 32) & 0x01FFFFFFu) | 0x80000000u;  // addr[56:32]|type=2
  int32x8 g1;
  g1[0] = (int)((1u << 16)        // data_size = 2 bytes
              | (1u << 20)        // pad_enable
              | (5u << 22)        // pad_interval: 2^(5+1)=64 DWORDs
              | (3u << 25));      // pad_amount: 3+1=4 DWORDs
  g1[1] = (int)((128u & 0xFFFFu) << 16);                // tensor_dim0 = 128 (lo16)
  g1[2] = (int)(((128u >> 16) & 0xFFFFu)                // tensor_dim0 (hi16)
              | ((tensor_rows & 0xFFFFu) << 16));       // tensor_dim1 (lo16)
  g1[3] = (int)(((tensor_rows >> 16) & 0xFFFFu)         // tensor_dim1 (hi16)
              | (128u << 16));                          // tile_dim0 = 128
  g1[4] = (int)(tile_rows & 0xFFFFu);                   // tile_dim1, tile_dim2=0
  g1[5] = (int)row_stride;                              // tensor_dim0_stride lo32
  g1[6] = 0;                                            // stride hi16, dim1_stride lo
  g1[7] = 0;
  int32x4 z4 = {0, 0, 0, 0};
  int32x8 z8 = {0, 0, 0, 0, 0, 0, 0, 0};
  // 6-arg variant (clang-23 / therock-10.0): (g0, g1, g2, g3, g4, cpol)
  __builtin_amdgcn_tensor_load_to_lds(g0, g1, z4, z4, z8, 0);
}

// ---------------------------------------------------------------------------
// Kernel 1: q = x @ Wq^T  (M=B*T=16384, N=HEAD=128, K=C=2048), RoPE epilogue.
// Block = 256 threads (8 waves), 128x128 output tile, K-step 32.
// (fp32 -> bf16 conversion happens on the way into LDS, so no TDM here.)
// ---------------------------------------------------------------------------
#define LDA 40  // padded LDS stride (bf16 elems): 80B rows -> conflict-free b128

__global__ __launch_bounds__(256) void qproj_rope_kernel(
    const float* __restrict__ x, const float* __restrict__ wq,
    bf16_t* __restrict__ qrot) {
  __shared__ __align__(16) bf16_t As[128 * LDA];
  __shared__ __align__(16) bf16_t Bs[128 * LDA];
  __shared__ float Qs[128 * 128];

  int tid = threadIdx.x;
  int lane = tid & 31;
  int wave = tid >> 5;
  int m0 = blockIdx.x * 128;

  v8f acc[8];
  const v8f vzero = {0.f, 0.f, 0.f, 0.f, 0.f, 0.f, 0.f, 0.f};
#pragma unroll
  for (int i = 0; i < 8; ++i) acc[i] = vzero;

  for (int k0 = 0; k0 < C_DIM; k0 += 32) {
    for (int i = tid; i < 128 * 32; i += 256) {
      int r = i >> 5, c = i & 31;
      As[r * LDA + c] = f2bf(x[(size_t)(m0 + r) * C_DIM + k0 + c]);
      Bs[r * LDA + c] = f2bf(wq[(size_t)r * C_DIM + k0 + c]);
    }
    __syncthreads();
    v16bf a = load_fragA(As, wave * 16, 0, LDA, lane);
#pragma unroll
    for (int nt = 0; nt < 8; ++nt) {
      v16bf bq = load_fragB(Bs, nt * 16, 0, LDA, lane);
      acc[nt] = wmma_bf16(a, bq, acc[nt]);
    }
    __syncthreads();
  }

  // stage fp32 q tile to LDS for the cross-lane RoPE pairing
  int hi = (lane < 16) ? 0 : 8;
  int nlo = lane & 15;
#pragma unroll
  for (int nt = 0; nt < 8; ++nt)
#pragma unroll
    for (int r = 0; r < 8; ++r)
      Qs[(wave * 16 + r + hi) * 128 + nt * 16 + nlo] = acc[nt][r];
  __syncthreads();

  // RoPE: 2 threads per row, 32 pairs each
  int row = tid >> 1;
  int gm = m0 + row;
  int t = gm & (T_SEQ - 1);
  const float* qr = &Qs[row * 128];
  size_t ob = (size_t)gm * HEAD;
  int i0 = (tid & 1) * 32;
  for (int i = i0; i < i0 + 32; ++i) {
    float inv = __powf(10000.f, -(2.f * (float)i) / 128.f);
    float ang = (float)t * inv;
    float c, s;
    __sincosf(ang, &c, &s);
    float x1 = qr[2 * i], x2 = qr[2 * i + 1];
    qrot[ob + 2 * i]     = f2bf(x1 * c - x2 * s);
    qrot[ob + 2 * i + 1] = f2bf(x1 * s + x2 * c);
  }
}

// ---------------------------------------------------------------------------
// Kernel 2: RoPE(key) -> bf16, value -> bf16.  One thread per rotation pair.
// ---------------------------------------------------------------------------
__global__ __launch_bounds__(256) void rope_kv_kernel(
    const float* __restrict__ key, const float* __restrict__ val,
    bf16_t* __restrict__ krot, bf16_t* __restrict__ vb) {
  int idx = blockIdx.x * 256 + threadIdx.x;  // [0, B*T*HEAD/2)
  int row = idx >> 6;                        // B*T row
  int i = idx & 63;                          // pair index
  int t = row & (T_SEQ - 1);
  float inv = __powf(10000.f, -(2.f * (float)i) / 128.f);
  float ang = (float)t * inv;
  float c, s;
  __sincosf(ang, &c, &s);
  size_t base = (size_t)row * HEAD + 2 * i;
  float k1 = key[base], k2 = key[base + 1];
  krot[base]     = f2bf(k1 * c - k2 * s);
  krot[base + 1] = f2bf(k1 * s + k2 * c);
  vb[base]       = f2bf(val[base]);
  vb[base + 1]   = f2bf(val[base + 1]);
}

// ---------------------------------------------------------------------------
// Kernel 3: causal flash attention.
// Block = 256 threads (8 waves).  128 query rows/block (wave owns 16 rows),
// 64 KV columns per iteration, online softmax in fp32.
// Q tile (once) and K tiles (per iter) DMA'd into padded LDS by the TDM.
// ---------------------------------------------------------------------------
#define FA_BM 128
#define FA_BN 64
#define LQ 136  // 272B rows: TDM pad (64+4 DWORDs), 16B-aligned frag loads
#define LK 136
#define LV 72   // 144B rows
#define LP 72
#define LSS 65  // fp32 scores: stride 65 -> bank = (row + j) % 64

__global__ __launch_bounds__(256) void flash_attn_kernel(
    const bf16_t* __restrict__ qrot, const bf16_t* __restrict__ krot,
    const bf16_t* __restrict__ vb, float* __restrict__ out) {
  __shared__ __align__(16) bf16_t Qs[FA_BM * LQ];
  __shared__ __align__(16) bf16_t Ks[FA_BN * LK];
  __shared__ __align__(16) bf16_t Vt[HEAD * LV];   // V transposed: Vt[d][s]
  __shared__ __align__(16) bf16_t Ps[FA_BM * LP];  // probabilities (bf16)
  __shared__ float Ss[FA_BM * LSS];                // raw scores (fp32)
  __shared__ float mS[FA_BM], lS[FA_BM], aS[FA_BM];

  int tid = threadIdx.x;
  int lane = tid & 31;
  int wave = tid >> 5;
  int iq = blockIdx.x & (T_SEQ / FA_BM - 1);  // 32 q-tiles per batch
  int b  = blockIdx.x >> 5;
  size_t rowbase = (size_t)b * T_SEQ + (size_t)iq * FA_BM;

  // TDM: DMA the whole Q tile (128 x 128 bf16) into padded LDS (once).
  if (wave == 0) {
    tdm_load_tile_bf16(lds_off(Qs), qrot + rowbase * HEAD, FA_BM,
                       (unsigned)T_SEQ, HEAD);
  }
  if (tid < FA_BM) { mS[tid] = -__builtin_inff(); lS[tid] = 0.f; }

  int mrow = wave * 16;
  int hi = (lane < 16) ? 0 : 8;
  int nlo = lane & 15;
  const v8f vzero = {0.f, 0.f, 0.f, 0.f, 0.f, 0.f, 0.f, 0.f};
  v8f oacc[8];
#pragma unroll
  for (int i = 0; i < 8; ++i) oacc[i] = vzero;

  int ntile = (iq + 1) * (FA_BM / FA_BN);  // causal early exit
  for (int js = 0; js < ntile; ++js) {
    int s0 = js * FA_BN;
    size_t sbase = (size_t)b * T_SEQ + s0;
    __syncthreads();  // prev iteration's Ks/Vt/Ps readers are done
    // TDM: DMA K tile (64 x 128 bf16) into padded LDS.
    if (wave == 0) {
      tdm_load_tile_bf16(lds_off(Ks), krot + sbase * HEAD, FA_BN,
                         (unsigned)T_SEQ, HEAD);
    }
    for (int i = tid; i < FA_BN * HEAD; i += 256) {  // V transpose into LDS
      int s = i >> 7, d = i & 127;
      Vt[d * LV + s] = vb[(sbase + s) * HEAD + d];
    }
    if (wave == 0) __builtin_amdgcn_s_wait_tensorcnt(0);  // Q (1st iter) + K
    __syncthreads();

    // S = Q K^T  (wave: 16 rows x 64 cols, K-dim = HEAD = 4 wmma steps)
    v8f sacc[4];
#pragma unroll
    for (int i = 0; i < 4; ++i) sacc[i] = vzero;
#pragma unroll
    for (int kk = 0; kk < 4; ++kk) {
      v16bf a = load_fragA(Qs, mrow, kk * 32, LQ, lane);
#pragma unroll
      for (int nt = 0; nt < 4; ++nt) {
        v16bf bk = load_fragB(Ks, nt * 16, kk * 32, LK, lane);
        sacc[nt] = wmma_bf16(a, bk, sacc[nt]);
      }
    }
    const float scale = 0.08838834764831845f;  // HEAD^-0.5
#pragma unroll
    for (int nt = 0; nt < 4; ++nt)
#pragma unroll
      for (int r = 0; r < 8; ++r)
        Ss[(mrow + r + hi) * LSS + nt * 16 + nlo] = sacc[nt][r] * scale;
    __syncthreads();

    // online softmax: one thread per query row
    if (tid < FA_BM) {
      int row = tid;
      int t = iq * FA_BM + row;
      float m_old = mS[row];
      float rmax = -1e30f;
#pragma unroll 8
      for (int j = 0; j < FA_BN; ++j) {
        float v = Ss[row * LSS + j];
        if (s0 + j > t) v = -1e30f;
        rmax = fmaxf(rmax, v);
      }
      float m_new = fmaxf(m_old, rmax);
      float alpha = __expf(m_old - m_new);
      float sum = 0.f;
#pragma unroll 8
      for (int j = 0; j < FA_BN; ++j) {
        float v = Ss[row * LSS + j];
        if (s0 + j > t) v = -1e30f;
        float p = __expf(v - m_new);
        sum += p;
        Ps[row * LP + j] = f2bf(p);
      }
      lS[row] = lS[row] * alpha + sum;
      mS[row] = m_new;
      aS[row] = alpha;
    }
    __syncthreads();

    // rescale O, then O += P V  (K-dim = 64 = 2 wmma steps, 8 d-tiles)
#pragma unroll
    for (int r = 0; r < 8; ++r) {
      float al = aS[mrow + r + hi];
#pragma unroll
      for (int nt = 0; nt < 8; ++nt) oacc[nt][r] *= al;
    }
#pragma unroll
    for (int kk = 0; kk < 2; ++kk) {
      v16bf a = load_fragA(Ps, mrow, kk * 32, LP, lane);
#pragma unroll
      for (int nt = 0; nt < 8; ++nt) {
        v16bf bv = load_fragB(Vt, nt * 16, kk * 32, LV, lane);
        oacc[nt] = wmma_bf16(a, bv, oacc[nt]);
      }
    }
  }
  __syncthreads();

  // finalize: O /= l, write fp32
#pragma unroll
  for (int r = 0; r < 8; ++r) {
    int row = mrow + r + hi;
    float linv = 1.f / lS[row];
    size_t ob = (rowbase + row) * HEAD;
#pragma unroll
    for (int nt = 0; nt < 8; ++nt)
      out[ob + nt * 16 + nlo] = oacc[nt][r] * linv;
  }
}

// ---------------------------------------------------------------------------
extern "C" void kernel_launch(void* const* d_in, const int* in_sizes, int n_in,
                              void* d_out, int out_size, void* d_ws,
                              size_t ws_size, hipStream_t stream) {
  (void)in_sizes; (void)n_in; (void)out_size; (void)ws_size;
  const float* x   = (const float*)d_in[0];
  const float* key = (const float*)d_in[1];
  const float* val = (const float*)d_in[2];
  const float* wq  = (const float*)d_in[3];
  float* out = (float*)d_out;

  const size_t NROW = (size_t)BATCH * T_SEQ;  // 16384
  bf16_t* qrot = (bf16_t*)d_ws;               // 4 MB
  bf16_t* krot = qrot + NROW * HEAD;          // 4 MB
  bf16_t* vbf  = krot + NROW * HEAD;          // 4 MB

  qproj_rope_kernel<<<dim3((unsigned)(NROW / 128)), 256, 0, stream>>>(x, wq, qrot);
  rope_kv_kernel<<<dim3((unsigned)(NROW * (HEAD / 2) / 256)), 256, 0, stream>>>(
      key, val, krot, vbf);
  flash_attn_kernel<<<dim3((unsigned)(BATCH * (T_SEQ / FA_BM))), 256, 0, stream>>>(
      qrot, krot, vbf, out);
}
